// BayesLSTM_32263794328029
// MI455X (gfx1250) — compile-verified, matrix operand order checked by reading, not run on hardware
//
#include <hip/hip_runtime.h>
#include <stdint.h>

// ---------------------------------------------------------------------------
// Bayesian LSTM for MI455X (gfx1250, wave32, WMMA bf16 16x16x32).
// B=128, L=256, D=512, H=1024. Weights resampled per time step and generated
// on-the-fly into LDS (never touch HBM); params stay L2-resident (48MB << 192MB).
// One kernel launch per time step handles the recurrent dependency.
// Double-buffered LDS panels: one workgroup barrier per K-panel; weight
// sampling (VALU/trans + global mu loads) overlaps the WMMA pipeline; the
// activation panel is moved global->LDS with GLOBAL_LOAD_ASYNC_TO_LDS_B128
// (ASYNCcnt) so the generator never stalls on loadcnt.
// ---------------------------------------------------------------------------

#define B_      128
#define L_      256
#define D_      512
#define H_      1024
#define FEATS   32            // h-features per workgroup
#define NCOL    128           // 4 gates * FEATS columns of the gates tile
#define THREADS 512           // 16 wave32s

typedef __attribute__((ext_vector_type(16))) __bf16 v16bf;
typedef __attribute__((ext_vector_type(8)))  float  v8f;

// ----------------------------- RNG helpers --------------------------------

__device__ __forceinline__ uint32_t hash32(uint32_t x) {
  x ^= x >> 16; x *= 0x7feb352du;
  x ^= x >> 15; x *= 0x846ca68bu;
  x ^= x >> 16;
  return x;
}

// Two standard normals from a stateless counter (Box-Muller).
__device__ __forceinline__ float2 gauss2(uint32_t key, uint32_t idx) {
  uint32_t a = hash32(key ^ hash32(idx ^ 0x9e3779b9u));
  uint32_t b = hash32(a ^ 0x68e31da4u);
  float u1 = ((a >> 9) + 0.5f) * (1.0f / 8388608.0f);  // (0,1)
  float u2 = ((b >> 9) + 0.5f) * (1.0f / 8388608.0f);
  float r = __fsqrt_rn(-2.0f * __logf(u1));
  float s, c;
  __sincosf(6.28318530717958647692f * u2, &s, &c);
  return make_float2(r * c, r * s);
}

__device__ __forceinline__ float softplus_f(float x) {
  return __logf(1.0f + __expf(x));
}

__device__ __forceinline__ float sigmoid_f(float x) {
  return 1.0f / (1.0f + __expf(-x));
}

__device__ __forceinline__ float tanh_f(float x) {
  float e = __expf(-2.0f * fabsf(x));
  float r = (1.0f - e) / (1.0f + e);
  return copysignf(r, x);
}

__device__ __forceinline__ unsigned short f2bf(float f) {
  union { float f; uint32_t u; } v; v.f = f;
  uint32_t r = v.u + 0x7fffu + ((v.u >> 16) & 1u);   // round-to-nearest-even
  return (unsigned short)(r >> 16);
}

// ----------------------- async global->LDS helpers -------------------------
// GLOBAL_LOAD_ASYNC_TO_LDS_B128: VDST = LDS byte address (low 32 bits of the
// flat LDS-aperture address per ISA 10.2), VADDR = 64-bit global address.
// Tracked by ASYNCcnt; producer waits ASYNCcnt==0 before signaling barrier.

__device__ __forceinline__ void async_g2l_b128(void* lds_dst, const void* gsrc) {
  uint32_t lds = (uint32_t)(uintptr_t)lds_dst;
  unsigned long long ga = (unsigned long long)(uintptr_t)gsrc;
  asm volatile("global_load_async_to_lds_b128 %0, %1, off"
               :: "v"(lds), "v"(ga) : "memory");
}

__device__ __forceinline__ void wait_async0() {
  asm volatile("s_wait_asynccnt 0x0" ::: "memory");
}

// ----------------------------- shared memory -------------------------------
// Panels (double-buffered) and the epilogue gates buffer are not live at the
// same time -> overlap them. Total static LDS = 32 KB.

union __attribute__((aligned(32))) Smem {
  struct {
    unsigned short A[2][B_ * 32];     // 2 x 8 KB  (x_t / h panel, bf16)
    unsigned short W[2][NCOL * 32];   // 2 x 8 KB  (sampled weights, bf16)
  } pan;
  float gates[64 * NCOL];             // 32 KB (epilogue exchange)
};

// ------------------- panel generator (A load + W sample) -------------------

__device__ __forceinline__ void gen_panel(
    unsigned short* __restrict__ lA, unsigned short* __restrict__ lW,
    const unsigned short* __restrict__ arow,
    const float* __restrict__ mup, const float* __restrict__ lsp,
    uint32_t key, int wrow, int kdim, int k0, int rowv, int kk)
{
  // A panel: 128x32 bf16, DMA'd straight into LDS (no VGPR staging).
  async_g2l_b128(&lA[rowv * 32 + kk], &arow[k0 + kk]);

  // W panel: 8 sampled weights per thread: mu + softplus(ls) * eps
  __builtin_prefetch(&mup[k0 + 32 + kk], 0, 1);   // next panel's params
  __builtin_prefetch(&lsp[k0 + 32 + kk], 0, 1);
  union { unsigned short us[8]; uint4 q; } wv;
  #pragma unroll
  for (int i = 0; i < 8; i += 2) {
    uint32_t idx = (uint32_t)(wrow * kdim + k0 + kk + i);
    float2 e = gauss2(key, idx);
    float s0 = softplus_f(lsp[k0 + kk + i]);
    float s1 = softplus_f(lsp[k0 + kk + i + 1]);
    wv.us[i]     = f2bf(fmaf(s0, e.x, mup[k0 + kk + i]));
    wv.us[i + 1] = f2bf(fmaf(s1, e.y, mup[k0 + kk + i + 1]));
  }
  *(uint4*)&lW[rowv * 32 + kk] = wv.q;
}

// -------------------- fused sample-weights + WMMA phase --------------------
// Accumulates gates[128 x 128] += A[128 x kdim] * sampled_W^T[kdim x 128].

__device__ __forceinline__ void mm_phase(
    const unsigned short* __restrict__ Aptr, int astride, int kdim,
    const float* __restrict__ mu, const float* __restrict__ ls,
    uint32_t key, int f0, Smem& sm, v8f acc[4])
{
  const int tid  = threadIdx.x;
  const int wave = tid >> 5;
  const int lane = tid & 31;
  const int nt   = wave & 7;        // N-tile (16 cols) owned by this wave
  const int mg   = wave >> 3;       // 0/1 -> which 4 M-tiles
  const int rowv = tid >> 2;        // 0..127: panel row for cooperative work
  const int kk   = (tid & 3) * 8;   // 0,8,16,24: 8 contiguous k per thread

  const int gate = rowv >> 5;
  const int feat = f0 + (rowv & 31);
  const int wrow = gate * H_ + feat;                 // row in [4H, kdim] weight
  const float* mup = mu + (size_t)wrow * kdim;
  const float* lsp = ls + (size_t)wrow * kdim;
  const unsigned short* arow = Aptr + (size_t)rowv * astride;

  // Prime buffer 0.
  gen_panel(sm.pan.A[0], sm.pan.W[0], arow, mup, lsp, key, wrow, kdim, 0, rowv, kk);

  for (int k0 = 0; k0 < kdim; k0 += 32) {
    const int cur = (k0 >> 5) & 1;
    wait_async0();                   // this wave's async LDS writes landed
    __syncthreads();                 // panel `cur` ready; `cur^1` free to fill

    // WMMA first so its LDS waits don't serialize on the generator's VMEM.
    const unsigned short* lA = sm.pan.A[cur];
    const unsigned short* lW = sm.pan.W[cur];
    const v16bf bfrag =
        *(const v16bf*)&lW[(nt * 16 + (lane & 15)) * 32 + (lane >> 4) * 16];
    #pragma unroll
    for (int i = 0; i < 4; ++i) {
      const int mt = mg * 4 + i;
      const v16bf afrag =
          *(const v16bf*)&lA[(mt * 16 + (lane & 15)) * 32 + (lane >> 4) * 16];
      acc[i] = __builtin_amdgcn_wmma_f32_16x16x32_bf16(
          false, afrag, false, bfrag, (short)0, acc[i], false, false);
    }

    // Generate the next panel into the other buffer while WMMAs drain.
    if (k0 + 32 < kdim)
      gen_panel(sm.pan.A[cur ^ 1], sm.pan.W[cur ^ 1],
                arow, mup, lsp, key, wrow, kdim, k0 + 32, rowv, kk);
  }
  __syncthreads();                   // last panel's WMMA reads complete
}

// ------------------------------ step kernel --------------------------------

__global__ __launch_bounds__(THREADS)
void bayes_lstm_step(const float* __restrict__ w_mu, const float* __restrict__ w_ls,
                     const float* __restrict__ wb_mu, const float* __restrict__ wb_ls,
                     const float* __restrict__ u_mu, const float* __restrict__ u_ls,
                     const float* __restrict__ ub_mu, const float* __restrict__ ub_ls,
                     const unsigned short* __restrict__ xbf,
                     unsigned short* __restrict__ hbf,
                     float* __restrict__ cst,
                     float* __restrict__ out, int t)
{
  __shared__ Smem sm;

  const int tid  = threadIdx.x;
  const int wave = tid >> 5;
  const int lane = tid & 31;
  const int f0   = blockIdx.x * FEATS;

  const uint32_t tk    = (uint32_t)t * 0x9e3779b9u;
  const uint32_t keyW  = hash32(tk ^ 0x1b873593u);
  const uint32_t keyU  = hash32(tk ^ 0x85ebca6bu);
  const uint32_t keyWB = hash32(tk ^ 0xc2b2ae35u);
  const uint32_t keyUB = hash32(tk ^ 0x27d4eb2fu);

  v8f acc[4];
  {
    v8f z = {};
    acc[0] = z; acc[1] = z; acc[2] = z; acc[3] = z;
  }

  // gates += x_t @ W_t^T  (K = D), then gates += h @ U_t^T (K = H)
  mm_phase(xbf + (size_t)t * B_ * D_, D_, D_, w_mu, w_ls, keyW, f0, sm, acc);
  mm_phase(hbf, H_, H_, u_mu, u_ls, keyU, f0, sm, acc);

  // Per-thread sampled bias for its feature column j (fixed across passes).
  const int j = tid & 31;
  float bias[4];
  #pragma unroll
  for (int g = 0; g < 4; ++g) {
    int bi = g * H_ + f0 + j;
    float2 e0 = gauss2(keyWB, (uint32_t)bi);
    float2 e1 = gauss2(keyUB, (uint32_t)bi);
    bias[g] = wb_mu[bi] + softplus_f(wb_ls[bi]) * e0.x
            + ub_mu[bi] + softplus_f(ub_ls[bi]) * e1.x;
  }

  // Epilogue in two 64-batch-row halves via the (overlapped) LDS gates buffer.
  const int nt = wave & 7;
  for (int pass = 0; pass < 2; ++pass) {
    if ((wave >> 3) == pass) {
      #pragma unroll
      for (int i = 0; i < 4; ++i) {
        const int mbase = i * 16 + ((lane < 16) ? 0 : 8);
        const int ncol  = nt * 16 + (lane & 15);
        #pragma unroll
        for (int r = 0; r < 8; ++r)
          sm.gates[(mbase + r) * NCOL + ncol] = acc[i][r];
      }
    }
    __syncthreads();

    #pragma unroll
    for (int q = 0; q < 4; ++q) {
      const int bl = (tid >> 5) * 4 + q;        // 0..63 local batch row
      const int b  = pass * 64 + bl;
      float gf = sigmoid_f(sm.gates[bl * NCOL +      j] + bias[0]);  // forget
      float gi = sigmoid_f(sm.gates[bl * NCOL + 32 + j] + bias[1]);  // input
      float go = sigmoid_f(sm.gates[bl * NCOL + 64 + j] + bias[2]);  // output
      float ch = tanh_f   (sm.gates[bl * NCOL + 96 + j] + bias[3]);  // candidate
      const int hidx = b * H_ + f0 + j;
      float cn = gf * cst[hidx] + gi * ch;
      float hn = go * tanh_f(cn);
      cst[hidx] = cn;
      hbf[hidx] = f2bf(hn);
      out[((size_t)b * L_ + t) * H_ + f0 + j] = hn;
    }
    __syncthreads();
  }
}

// ------------------------------ small kernels ------------------------------

// x [B,L,D] f32 -> time-major bf16 [L,B,D] (WMMA A operand layout).
__global__ void xcast_kernel(const float* __restrict__ x,
                             unsigned short* __restrict__ xbf) {
  unsigned o = blockIdx.x * blockDim.x + threadIdx.x;   // < L*B*D
  unsigned t = o / (B_ * D_);
  unsigned r = o % (B_ * D_);
  unsigned b = r / D_;
  unsigned d = r % D_;
  xbf[o] = f2bf(x[((size_t)b * L_ + t) * D_ + d]);
}

__global__ void init_kernel(unsigned short* __restrict__ hbf,
                            float* __restrict__ cst) {
  int i = blockIdx.x * blockDim.x + threadIdx.x;        // < B*H
  hbf[i] = 0;
  cst[i] = 0.0f;
}

// Tuple tail: h_last = outputs[:, L-1, :], c_last = c state.
__global__ void tail_kernel(float* __restrict__ out,
                            const float* __restrict__ cst) {
  int i = blockIdx.x * blockDim.x + threadIdx.x;        // < B*H
  int b = i >> 10;
  int jj = i & (H_ - 1);
  size_t base = (size_t)B_ * L_ * H_;
  out[base + i] = out[((size_t)b * L_ + (L_ - 1)) * H_ + jj];
  out[base + (size_t)B_ * H_ + i] = cst[i];
}

// -------------------------------- launcher ---------------------------------

extern "C" void kernel_launch(void* const* d_in, const int* in_sizes, int n_in,
                              void* d_out, int out_size, void* d_ws, size_t ws_size,
                              hipStream_t stream) {
  (void)in_sizes; (void)n_in; (void)out_size; (void)ws_size;

  const float* x     = (const float*)d_in[0];
  const float* w_mu  = (const float*)d_in[1];
  const float* w_ls  = (const float*)d_in[2];
  const float* wb_mu = (const float*)d_in[3];
  const float* wb_ls = (const float*)d_in[4];
  const float* u_mu  = (const float*)d_in[5];
  const float* u_ls  = (const float*)d_in[6];
  const float* ub_mu = (const float*)d_in[7];
  const float* ub_ls = (const float*)d_in[8];

  // Workspace: [xbf: L*B*D bf16 = 32MB][hbf: B*H bf16][c: B*H f32]  (~34 MB)
  unsigned short* xbf = (unsigned short*)d_ws;
  unsigned short* hbf = xbf + (size_t)L_ * B_ * D_;
  float*          cst = (float*)(hbf + (size_t)B_ * H_);
  float*          out = (float*)d_out;

  xcast_kernel<<<(L_ * B_ * D_) / THREADS, THREADS, 0, stream>>>(x, xbf);
  init_kernel<<<(B_ * H_) / THREADS, THREADS, 0, stream>>>(hbf, cst);

  for (int t = 0; t < L_; ++t) {
    bayes_lstm_step<<<H_ / FEATS, THREADS, 0, stream>>>(
        w_mu, w_ls, wb_mu, wb_ls, u_mu, u_ls, ub_mu, ub_ls,
        xbf, hbf, cst, out, t);
  }

  tail_kernel<<<(B_ * H_) / THREADS, THREADS, 0, stream>>>(out, cst);
}
